// DistLayer_88794153877507
// MI455X (gfx1250) — matrix operand order repaired and checked by reading, not compiled, not run on hardware
//
#include <hip/hip_runtime.h>

// ---------------------------------------------------------------------------
// Problem constants (match reference)
// ---------------------------------------------------------------------------
constexpr int N_ROWS   = 1000000;     // pair rows
constexpr int D_IN     = 192;         // fc1 input dim
constexpr int N_AE     = 64;          // fc1 output dim
constexpr int D_OUT    = 128;         // concat output dim
constexpr int NUM_SEG  = 50000;
constexpr int ROW_TILES = N_ROWS / 16;      // 62500 (exact)
constexpr int LDS_K_STRIDE = 200;           // padded halfs per column row (bank-conflict free)
constexpr int BN_PART_BLOCKS = 512;

typedef __attribute__((ext_vector_type(16))) _Float16 v16h;
typedef __attribute__((ext_vector_type(8)))  _Float16 v8h;
typedef __attribute__((ext_vector_type(8)))  float    v8f;
typedef __attribute__((ext_vector_type(4)))  float    v4f;

// ---------------------------------------------------------------------------
// Kernel 1: h = x @ W + b   (pre-BN), written to out[:, 0:64]
// One wave computes a 16-row x 64-col strip via 24 x v_wmma_f32_16x16x32_f16.
// W is staged into LDS as f16, transposed [col][k] with padding so the 16
// lanes of a B-fragment hit 16 distinct banks. B fragments are re-read from
// LDS every row tile (an opaque zero offset defeats LICM) so the wave stays
// under ~80 VGPRs -> high occupancy for this HBM-streaming kernel.
// ---------------------------------------------------------------------------
__global__ void __launch_bounds__(256) gemm_bias_kernel(
    const float* __restrict__ x, const float* __restrict__ W,
    const float* __restrict__ bias, float* __restrict__ out)
{
    __shared__ _Float16 sW[N_AE * LDS_K_STRIDE];   // 25.6 KB

    // Stage W (f32, row-major [k][c]) -> LDS f16 transposed [c][k]
    for (int idx = threadIdx.x; idx < D_IN * N_AE; idx += 256) {
        const int k = idx >> 6;        // idx / 64  (coalesced global read)
        const int c = idx & 63;
        sW[c * LDS_K_STRIDE + k] = (_Float16)W[idx];
    }
    __syncthreads();

    const int lane    = threadIdx.x & 31;
    const int halfSel = lane >> 4;          // 0: lanes 0-15, 1: lanes 16-31
    const int l15     = lane & 15;
    const int waveId  = blockIdx.x * 8 + (threadIdx.x >> 5);
    const int numWaves = gridDim.x * 8;

    // bias per column tile (C/D layout: col = 16*t + (lane&15) for all 8 VGPRs)
    float bias_c[4];
#pragma unroll
    for (int t = 0; t < 4; ++t) bias_c[t] = bias[t * 16 + l15];

    // Opaque zero: the empty asm below "modifies" it each iteration, which
    // makes the LDS B-fragment addresses loop-variant and blocks LICM from
    // hoisting 24 fragments (192 VGPRs) out of the row-tile loop.
    unsigned sWoff = 0;

    for (int rt = waveId; rt < ROW_TILES; rt += numWaves) {
        asm volatile("" : "+v"(sWoff));
        const int row0 = rt * 16;
        const int arow = row0 + l15;

        // Prefetch next row tile for this wave (global_prefetch_b8)
        if (rt + numWaves < ROW_TILES) {
            const float* pf = x + (size_t)((rt + numWaves) * 16 + l15) * D_IN;
            __builtin_prefetch(pf, 0, 0);
        }

        v8f acc[4];
#pragma unroll
        for (int t = 0; t < 4; ++t)
#pragma unroll
            for (int r = 0; r < 8; ++r) acc[t][r] = bias_c[t];

#pragma unroll
        for (int ks = 0; ks < 6; ++ks) {
            const int k0 = ks * 32;

            // A fragment (16-bit A 16x32 layout):
            //   lane halfSel=0: K = k0+0..7   and k0+16..23
            //   lane halfSel=1: K = k0+8..15  and k0+24..31
            const float* ap = x + (size_t)arow * D_IN + k0 + halfSel * 8;
            const v4f a0 = *(const v4f*)(ap);
            const v4f a1 = *(const v4f*)(ap + 4);
            const v4f a2 = *(const v4f*)(ap + 16);
            const v4f a3 = *(const v4f*)(ap + 20);
            v16h afrag;
#pragma unroll
            for (int i = 0; i < 4; ++i) {
                afrag[i]      = (_Float16)a0[i];
                afrag[4 + i]  = (_Float16)a1[i];
                afrag[8 + i]  = (_Float16)a2[i];
                afrag[12 + i] = (_Float16)a3[i];
            }

#pragma unroll
            for (int t = 0; t < 4; ++t) {
                // B fragment (16-bit B 32x16 layout):
                //   lanes 0-15 : K = k0+0..7  and k0+16..23 at N = n0+lane
                //   lanes 16-31: K = k0+8..15 and k0+24..31 at N = n0+lane-16
                const int col = t * 16 + l15;
                const _Float16* bp =
                    &sW[col * LDS_K_STRIDE + k0 + halfSel * 8 + sWoff];
                const v8h b0 = *(const v8h*)(bp);
                const v8h b1 = *(const v8h*)(bp + 16);
                const v16h bfrag = __builtin_shufflevector(
                    b0, b1, 0, 1, 2, 3, 4, 5, 6, 7, 8, 9, 10, 11, 12, 13, 14, 15);

                acc[t] = __builtin_amdgcn_wmma_f32_16x16x32_f16(
                    /*neg_a=*/false, afrag, /*neg_b=*/false, bfrag,
                    /*c_mod=*/(short)0, acc[t],
                    /*reuse_a=*/false, /*reuse_b=*/false);
            }
        }

        // Store pre-BN h into out[:, 0:64].
        // C/D layout: VGPR r -> row r (lanes 0-15) / row r+8 (lanes 16-31).
#pragma unroll
        for (int t = 0; t < 4; ++t) {
            const int col = t * 16 + l15;
#pragma unroll
            for (int r = 0; r < 8; ++r) {
                out[(size_t)(row0 + r + 8 * halfSel) * D_OUT + col] = acc[t][r];
            }
        }
    }
}

// ---------------------------------------------------------------------------
// Kernel 2: per-block partial column sums / sumsq of h (deterministic)
// ---------------------------------------------------------------------------
__global__ void __launch_bounds__(256) bn_partial_kernel(
    const float* __restrict__ out, float* __restrict__ partial)
{
    const int c    = threadIdx.x & 63;
    const int rsub = threadIdx.x >> 6;       // 0..3
    float s = 0.f, s2 = 0.f;
    for (int row = blockIdx.x * 4 + rsub; row < N_ROWS; row += gridDim.x * 4) {
        const float v = out[(size_t)row * D_OUT + c];
        s += v;  s2 += v * v;
    }
    __shared__ float red[512];
    red[threadIdx.x]       = s;
    red[256 + threadIdx.x] = s2;
    __syncthreads();
    if (rsub == 0) {
        const float ts  = red[c] + red[64 + c] + red[128 + c] + red[192 + c];
        const float ts2 = red[256 + c] + red[320 + c] + red[384 + c] + red[448 + c];
        partial[blockIdx.x * 128 + c]      = ts;
        partial[blockIdx.x * 128 + 64 + c] = ts2;
    }
}

// ---------------------------------------------------------------------------
// Kernel 3: fold mean/var + gamma/beta into per-column scale & shift
// ---------------------------------------------------------------------------
__global__ void bn_finalize_kernel(
    const float* __restrict__ partial, int nparts,
    const float* __restrict__ gamma, const float* __restrict__ beta,
    float* __restrict__ stats)
{
    const int c = threadIdx.x;               // 0..63
    float s = 0.f, s2 = 0.f;
    for (int b = 0; b < nparts; ++b) {
        s  += partial[b * 128 + c];
        s2 += partial[b * 128 + 64 + c];
    }
    const float invN = 1.0f / (float)N_ROWS;
    const float mean = s * invN;
    const float var  = s2 * invN - mean * mean;     // biased variance (ddof=0)
    const float sc   = __frsqrt_rn(var + 1e-5f) * gamma[c];
    stats[c]      = sc;
    stats[64 + c] = beta[c] - mean * sc;
}

// ---------------------------------------------------------------------------
// Kernel 4: normalize + ReLU in place; segment-accumulate with run-length
// compression (ele_idx is sorted -> ~20 rows/segment -> ~16x fewer atomics)
// ---------------------------------------------------------------------------
__global__ void __launch_bounds__(256) bn_apply_pool_kernel(
    float* __restrict__ out, const int* __restrict__ ele,
    const float* __restrict__ stats,
    float* __restrict__ segSums, float* __restrict__ segCounts)
{
    const int c   = threadIdx.x & 63;
    const int sub = threadIdx.x >> 6;        // 0..3
    const int rstart = blockIdx.x * 256 + sub * 64;
    if (rstart >= N_ROWS) return;
    const int rend = min(rstart + 64, N_ROWS);

    const float sc = stats[c];
    const float sh = stats[64 + c];

    float runSum = 0.f, runCnt = 0.f;
    int runSeg = -1;
    for (int row = rstart; row < rend; ++row) {
        float v = out[(size_t)row * D_OUT + c];
        v = fmaxf(v * sc + sh, 0.f);
        out[(size_t)row * D_OUT + c] = v;
        const int seg = ele[row];
        if (seg != runSeg) {
            if (runSeg >= 0) {
                atomicAdd(&segSums[(size_t)runSeg * N_AE + c], runSum);
                if (c == 0) atomicAdd(&segCounts[runSeg], runCnt);
            }
            runSeg = seg; runSum = 0.f; runCnt = 0.f;
        }
        runSum += v; runCnt += 1.f;
    }
    if (runSeg >= 0) {
        atomicAdd(&segSums[(size_t)runSeg * N_AE + c], runSum);
        if (c == 0) atomicAdd(&segCounts[runSeg], runCnt);
    }
}

// ---------------------------------------------------------------------------
// Kernel 5: counts -> 1/max(count,1)
// ---------------------------------------------------------------------------
__global__ void __launch_bounds__(256) counts_inv_kernel(float* __restrict__ segCounts)
{
    const int i = blockIdx.x * 256 + threadIdx.x;
    if (i < NUM_SEG) segCounts[i] = 1.0f / fmaxf(segCounts[i], 1.0f);
}

// ---------------------------------------------------------------------------
// Kernel 6: out[:, 64:128] = pooled[ele_idx]  (fully coalesced writes)
// ---------------------------------------------------------------------------
__global__ void __launch_bounds__(256) gather_kernel(
    float* __restrict__ out, const int* __restrict__ ele,
    const float* __restrict__ segSums, const float* __restrict__ segInv)
{
    const long long total = (long long)N_ROWS * N_AE;
    for (long long idx = (long long)blockIdx.x * 256 + threadIdx.x;
         idx < total; idx += (long long)gridDim.x * 256) {
        const int row = (int)(idx >> 6);
        const int c   = (int)(idx & 63);
        const int seg = ele[row];
        out[(size_t)row * D_OUT + N_AE + c] =
            segSums[(size_t)seg * N_AE + c] * segInv[seg];
    }
}

// ---------------------------------------------------------------------------
// Orchestration
// ---------------------------------------------------------------------------
extern "C" void kernel_launch(void* const* d_in, const int* in_sizes, int n_in,
                              void* d_out, int out_size, void* d_ws, size_t ws_size,
                              hipStream_t stream)
{
    const float* x     = (const float*)d_in[0];
    const int*   ele   = (const int*)  d_in[2];
    const float* W     = (const float*)d_in[3];
    const float* bias  = (const float*)d_in[4];
    const float* gamma = (const float*)d_in[5];
    const float* beta  = (const float*)d_in[6];
    float* out = (float*)d_out;

    // Workspace layout (all 256B aligned)
    char* ws = (char*)d_ws;
    float* segSums   = (float*)(ws);                         // 50000*64*4 = 12,800,000
    float* segCounts = (float*)(ws + 12800000);              // 50000*4    =    200,000
    float* bnPartial = (float*)(ws + 13000192);              // 512*128*4  =    262,144
    float* bnStats   = (float*)(ws + 13000192 + 262144);     // 128*4

    // Atomically-accumulated buffers must be zero each call (part of the graph).
    hipMemsetAsync(ws, 0, 13000000, stream);

    gemm_bias_kernel<<<2048, 256, 0, stream>>>(x, W, bias, out);
    bn_partial_kernel<<<BN_PART_BLOCKS, 256, 0, stream>>>(out, bnPartial);
    bn_finalize_kernel<<<1, 64, 0, stream>>>(bnPartial, BN_PART_BLOCKS, gamma, beta, bnStats);
    bn_apply_pool_kernel<<<(N_ROWS + 255) / 256, 256, 0, stream>>>(out, ele, bnStats,
                                                                   segSums, segCounts);
    counts_inv_kernel<<<(NUM_SEG + 255) / 256, 256, 0, stream>>>(segCounts);
    gather_kernel<<<4096, 256, 0, stream>>>(out, ele, segSums, segCounts);
}
